// ConvLSTM_63617055588469
// MI455X (gfx1250) — matrix operand order repaired and checked by reading, not compiled, not run on hardware
//
#include <hip/hip_runtime.h>

typedef __bf16  v16bf __attribute__((ext_vector_type(16)));
typedef __bf16  v8bf  __attribute__((ext_vector_type(8)));
typedef __bf16  v4bf  __attribute__((ext_vector_type(4)));
typedef float   v8f   __attribute__((ext_vector_type(8)));

#define HC 32
#define B_  16
#define T_  64
#define HW  32
#define OC  128   // 4*HC gate channels
#define SP  1024  // 32*32 spatial
#define KD  32768 // HC*H*W feature dim

// ---------------- init: zero h (both buffers) and c ----------------
__global__ __launch_bounds__(256) void init_state(unsigned int* hA, unsigned int* hB, unsigned int* c) {
    int i = blockIdx.x * 256 + threadIdx.x;
    if (i < B_ * SP * HC / 2) { hA[i] = 0u; hB[i] = 0u; }   // bf16 pairs
    if (i < B_ * SP * HC)     { c[i]  = 0u; }               // fp32
}

// ---------------- weight prep: swizzle conv weights ----------------
// wB[tap][ntile(8)][lane(32)][k(16)] : per-lane contiguous 16 bf16 B-fragment
__global__ __launch_bounds__(256) void prep_weights(const float* __restrict__ conv_w,
                                                    __bf16* __restrict__ wB,
                                                    float* __restrict__ wX) {
    int e = blockIdx.x * 256 + threadIdx.x;
    const int NB = 25 * 8 * 32 * 16;
    if (e < NB) {
        int k     = e & 15;
        int lane  = (e >> 4) & 31;
        int ntile = (e >> 9) & 7;
        int tap   = e >> 12;
        int ky = tap / 5, kx = tap % 5;
        int oc = ntile * 16 + (lane & 15);
        int hc = ((lane & 16) >> 1) + (k & 7) + ((k & 8) << 1);
        int ic = 1 + hc;
        wB[e] = (__bf16)conv_w[((oc * 33 + ic) * 5 + ky) * 5 + kx];
    } else if (e < NB + OC * 25) {
        int r = e - NB;
        int oc = r / 25, tap = r % 25;
        int ky = tap / 5, kx = tap % 5;
        wX[r] = conv_w[((oc * 33 + 0) * 5 + ky) * 5 + kx];
    }
}

// ---------------- fp32 -> bf16 convert of out_w (row-major [o][d]) ----------
__global__ __launch_bounds__(256) void cvt_outw(const float* __restrict__ src,
                                                __bf16* __restrict__ dst) {
    size_t i = ((size_t)blockIdx.x * 256 + threadIdx.x) * 4;
    float4 v = *(const float4*)(src + i);
    v4bf o;
    o[0] = (__bf16)v.x; o[1] = (__bf16)v.y; o[2] = (__bf16)v.z; o[3] = (__bf16)v.w;
    *(v4bf*)(dst + i) = o;
}

__device__ __forceinline__ float sigm(float x) { return 1.0f / (1.0f + __expf(-x)); }

// ---------------- one ConvLSTM timestep ----------------
// grid (B, 16): block handles batch b, image rows y0..y0+1 (64 spatial x 128 ch)
__global__ __launch_bounds__(256) void lstm_step(const float* __restrict__ x,
                                                 const float* __restrict__ conv_b,
                                                 const __bf16* __restrict__ wB,
                                                 const float* __restrict__ wX,
                                                 const __bf16* __restrict__ hIn,
                                                 __bf16* __restrict__ hOut,
                                                 float* __restrict__ cSt,
                                                 __bf16* __restrict__ feat,
                                                 int t) {
    __shared__ __bf16 ht[6][36][40];     // haloed h tile, ch-last, padded 32->40
    __shared__ float  xt[6][36];         // haloed x tile
    __shared__ float  zt[64][129];       // gates pre-activation, padded
    __shared__ __bf16 hstage[32][72];    // h_new transpose buffer, padded

    const int b = blockIdx.x, wy = blockIdx.y;
    const int y0 = wy * 2;
    const int tid = threadIdx.x;

    // --- load haloed h: one 16B chunk per thread-iter (cell = 64B) ---
    for (int idx = tid; idx < 6 * 36 * 4; idx += 256) {
        int q = idx & 3, cell = idx >> 2;
        int col = cell % 36, row = cell / 36;
        int gy = y0 - 2 + row, gx = col - 2;
        float4 v = make_float4(0.f, 0.f, 0.f, 0.f);
        if (gy >= 0 && gy < HW && gx >= 0 && gx < HW)
            v = *(const float4*)(hIn + ((size_t)((b * HW + gy) * HW + gx)) * HC + q * 8);
        *(float4*)((char*)&ht[row][col][0] + q * 16) = v;
    }
    // --- load haloed x_t ---
    for (int idx = tid; idx < 6 * 36; idx += 256) {
        int col = idx % 36, row = idx / 36;
        int gy = y0 - 2 + row, gx = col - 2;
        float v = 0.f;
        if (gy >= 0 && gy < HW && gx >= 0 && gx < HW)
            v = x[((size_t)(b * T_ + t) * HW + gy) * HW + gx];
        xt[row][col] = v;
    }
    __syncthreads();

    // --- implicit GEMM over 25 taps, K=32 hidden channels per tap ---
    const int wave = tid >> 5, lane = tid & 31;
    const int ntile = wave;
    const int mlane = lane & 15, hi = lane >> 4;
    const __bf16* wbase = wB + ((size_t)ntile * 32 + lane) * 16;
    v16bf bnext = *(const v16bf*)wbase;                 // tap 0 (pipelined)
    v8f acc[4] = {};
    for (int tap = 0; tap < 25; ++tap) {
        v16bf bcur = bnext;
        if (tap < 24) bnext = *(const v16bf*)(wbase + (size_t)(tap + 1) * 4096);
        int ky = tap / 5, kx = tap % 5;
#pragma unroll
        for (int mt = 0; mt < 4; ++mt) {
            int yl = (mt >> 1) + ky;
            int xp = (mt & 1) * 16 + mlane + kx;
            const v8bf a0 = *(const v8bf*)&ht[yl][xp][hi * 8];
            const v8bf a1 = *(const v8bf*)&ht[yl][xp][hi * 8 + 16];
            v16bf afrag = __builtin_shufflevector(a0, a1, 0,1,2,3,4,5,6,7,8,9,10,11,12,13,14,15);
            acc[mt] = __builtin_amdgcn_wmma_f32_16x16x32_bf16(
                false, afrag, false, bcur, (short)0, acc[mt], false, false);
        }
    }
#pragma unroll
    for (int mt = 0; mt < 4; ++mt)
#pragma unroll
        for (int r = 0; r < 8; ++r)
            zt[mt * 16 + r + hi * 8][ntile * 16 + mlane] = acc[mt][r];
    __syncthreads();

    // --- add x-channel contribution + bias ---
    {
        int oc = tid & 127;
        float wrow[25];
#pragma unroll
        for (int tp = 0; tp < 25; ++tp) wrow[tp] = wX[oc * 25 + tp];
        float bias = conv_b[oc];
        for (int o = tid; o < 64 * OC; o += 256) {
            int p = o >> 7;
            int yl = p >> 5, xp = p & 31;
            float a = zt[p][oc] + bias;
#pragma unroll
            for (int tp = 0; tp < 25; ++tp)
                a += wrow[tp] * xt[yl + tp / 5][xp + tp % 5];
            zt[p][oc] = a;
        }
    }
    __syncthreads();

    // --- LSTM pointwise gates, state update ---
    for (int g = tid; g < 64 * HC; g += 256) {
        int p = g >> 5, hc = g & 31;
        float iv = sigm(zt[p][hc]);
        float fv = sigm(zt[p][hc + HC]);
        float ov = sigm(zt[p][hc + 2 * HC]);
        float gv = tanhf(zt[p][hc + 3 * HC]);
        int gy = y0 + (p >> 5), gx = p & 31;
        size_t ci = ((size_t)((b * HW + gy) * HW + gx)) * HC + hc;
        float cn = fv * cSt[ci] + iv * gv;
        cSt[ci] = cn;
        float hn = ov * tanhf(cn);
        hOut[ci] = (__bf16)hn;
        hstage[hc][p] = (__bf16)hn;
    }
    __syncthreads();
    // --- coalesced feat writeback: 16B per thread ---
    {
        int hc = tid >> 3, q = tid & 7;
        size_t base = (size_t)(b * T_ + t) * KD + (size_t)hc * SP + y0 * HW;
        *(float4*)(feat + base + q * 8) = *(const float4*)&hstage[hc][q * 8];
    }
}

// ---------------- projection GEMM: y(1024x1024) = feat(1024xKD) * wO^T + b ----
// grid (16, 8): WG tile M=64 x N=128; async double-buffered LDS A, pipelined B
__global__ __launch_bounds__(256) void proj_gemm(const __bf16* __restrict__ feat,
                                                 const __bf16* __restrict__ wO,
                                                 const float* __restrict__ out_b,
                                                 float* __restrict__ y) {
    __shared__ __bf16 shA[2][64][136];   // padded: 272B rows -> conflict-free frags
    const int mb = blockIdx.x, nb = blockIdx.y;
    const int tid = threadIdx.x, wave = tid >> 5, lane = tid & 31;
    const int ntile = wave, mlane = lane & 15, hi = lane >> 4;
    const int ocol = nb * 128 + ntile * 16 + mlane;
    const size_t brow = (size_t)ocol * KD;

    auto stage = [&](int buf, int k0) {
#pragma unroll
        for (int c = tid; c < 1024; c += 256) {
            int row = c >> 4, kc = c & 15;
            unsigned lds = (unsigned)(size_t)&shA[buf][row][kc * 8];   // LDS offset = low 32 bits
            const __bf16* g = feat + (size_t)(mb * 64 + row) * KD + k0 + kc * 8;
            asm volatile("global_load_async_to_lds_b128 %0, %1, off"
                         :: "v"(lds), "v"(g) : "memory");
        }
    };

    stage(0, 0);
    v16bf bn[4];
#pragma unroll
    for (int kb = 0; kb < 4; ++kb)
        bn[kb] = *(const v16bf*)(wO + brow + kb * 32 + hi * 16);
    asm volatile("s_wait_asynccnt 0x0" ::: "memory");
    __syncthreads();

    v8f acc[4] = {};
    for (int k0 = 0; k0 < KD; k0 += 128) {
        int cur = (k0 >> 7) & 1;
        v16bf bf[4];
#pragma unroll
        for (int kb = 0; kb < 4; ++kb) bf[kb] = bn[kb];
        if (k0 + 128 < KD) {
            stage(cur ^ 1, k0 + 128);                       // async, overlaps WMMAs
#pragma unroll
            for (int kb = 0; kb < 4; ++kb)                   // pipeline next B chunk
                bn[kb] = *(const v16bf*)(wO + brow + (k0 + 128) + kb * 32 + hi * 16);
        }
#pragma unroll
        for (int kb = 0; kb < 4; ++kb) {
            int kk = kb * 32;
#pragma unroll
            for (int mt = 0; mt < 4; ++mt) {
                const v8bf a0 = *(const v8bf*)&shA[cur][mt * 16 + mlane][kk + hi * 16];
                const v8bf a1 = *(const v8bf*)&shA[cur][mt * 16 + mlane][kk + hi * 16 + 8];
                v16bf afrag = __builtin_shufflevector(a0, a1, 0,1,2,3,4,5,6,7,8,9,10,11,12,13,14,15);
                acc[mt] = __builtin_amdgcn_wmma_f32_16x16x32_bf16(
                    false, afrag, false, bf[kb], (short)0, acc[mt], false, false);
            }
        }
        asm volatile("s_wait_asynccnt 0x0" ::: "memory");
        __syncthreads();
    }
    float bias = out_b[ocol];
#pragma unroll
    for (int mt = 0; mt < 4; ++mt)
#pragma unroll
        for (int r = 0; r < 8; ++r)
            y[(size_t)(mb * 64 + mt * 16 + r + hi * 8) * 1024 + ocol] = acc[mt][r] + bias;
}

extern "C" void kernel_launch(void* const* d_in, const int* in_sizes, int n_in,
                              void* d_out, int out_size, void* d_ws, size_t ws_size,
                              hipStream_t stream) {
    const float* x      = (const float*)d_in[0]; // (16,64,32,32)
    const float* conv_w = (const float*)d_in[1]; // (128,33,5,5)
    const float* conv_b = (const float*)d_in[2]; // (128)
    const float* out_w  = (const float*)d_in[3]; // (1024,32768)
    const float* out_b  = (const float*)d_in[4]; // (1024)
    float* y = (float*)d_out;

    char* ws = (char*)d_ws;
    size_t o = 0;
    auto take = [&](size_t bytes) { size_t r = o; o = (o + bytes + 255) & ~(size_t)255; return r; };
    __bf16* wB   = (__bf16*)(ws + take(25 * 8 * 32 * 16 * sizeof(__bf16)));
    float*  wX   = (float*) (ws + take(OC * 25 * sizeof(float)));
    __bf16* hA   = (__bf16*)(ws + take((size_t)B_ * SP * HC * 2));
    __bf16* hB   = (__bf16*)(ws + take((size_t)B_ * SP * HC * 2));
    float*  cSt  = (float*) (ws + take((size_t)B_ * SP * HC * 4));
    __bf16* feat = (__bf16*)(ws + take((size_t)B_ * T_ * KD * 2));
    __bf16* wO   = (__bf16*)(ws + take((size_t)1024 * KD * 2));

    init_state<<<(B_ * SP * HC + 255) / 256, 256, 0, stream>>>(
        (unsigned int*)hA, (unsigned int*)hB, (unsigned int*)cSt);
    prep_weights<<<(25 * 8 * 32 * 16 + OC * 25 + 255) / 256, 256, 0, stream>>>(conv_w, wB, wX);
    cvt_outw<<<(int)(((size_t)1024 * KD / 4 + 255) / 256), 256, 0, stream>>>(out_w, wO);

    for (int t = 0; t < T_; ++t) {
        const __bf16* hIn  = (t & 1) ? hB : hA;
        __bf16*       hOut = (t & 1) ? hA : hB;
        lstm_step<<<dim3(B_, 16), 256, 0, stream>>>(x, conv_b, wB, wX, hIn, hOut, cSt, feat, t);
    }

    proj_gemm<<<dim3(16, 8), 256, 0, stream>>>(feat, wO, out_b, y);
}
